// LGNNConcatRAT_34248069218688
// MI455X (gfx1250) — compile-verified, matrix operand order checked by reading, not run on hardware
//
#include <hip/hip_runtime.h>

// ---------------------------------------------------------------- types
typedef _Float16 f16;
typedef __attribute__((ext_vector_type(16))) _Float16 v16h;
typedef __attribute__((ext_vector_type(8)))  float    v8f;

#define WMMA(a, b, c) __builtin_amdgcn_wmma_f32_16x16x32_f16( \
    false, (a), false, (b), (short)0, (c), false, false)

union FragU { v16h v; unsigned u[8]; f16 h[16]; };

// Load a 16x32 f16 fragment (A layout; also valid for B when fed the
// N-major transposed matrix) from a row-major [rows][ld] array.
// lane 0-15: m=lane, k 0..7 / 16..23 ; lane 16-31: m=lane-16, k 8..15 / 24..31
__device__ __forceinline__ v16h load_frag_f16(const f16* p, int ld, int row0,
                                              int k0, int lane) {
  int m = lane & 15;
  int kh = (lane >> 4) << 3;
  FragU r;
  const f16* base = p + (size_t)(row0 + m) * ld + (k0 + kh);
#pragma unroll
  for (int vv = 0; vv < 8; ++vv) {
    int kb = ((vv >> 2) << 4) + ((vv & 3) << 1);
    r.u[vv] = *(const unsigned*)(base + kb);
  }
  return r.v;
}

// Same fragment but sourced from f32 (converted on load).
__device__ __forceinline__ v16h load_frag_f32(const float* p, int ld, int row0,
                                              int k0, int lane) {
  int m = lane & 15;
  int kh = (lane >> 4) << 3;
  FragU r;
  const float* base = p + (size_t)(row0 + m) * ld + (k0 + kh);
#pragma unroll
  for (int vv = 0; vv < 8; ++vv) {
    int kb = ((vv >> 2) << 4) + ((vv & 3) << 1);
    r.h[2 * vv]     = (f16)base[kb];
    r.h[2 * vv + 1] = (f16)base[kb + 1];
  }
  return r.v;
}

// ------------------------------------------------------- elementwise utils
__global__ void k_fill(float* p, float v, size_t n) {
  size_t i = (size_t)blockIdx.x * blockDim.x + threadIdx.x;
  if (i < n) p[i] = v;
}

__global__ void k_cvt(const float* __restrict__ in, f16* __restrict__ out, size_t n) {
  size_t i = (size_t)blockIdx.x * blockDim.x + threadIdx.x;
  if (i < n) out[i] = (f16)in[i];
}

// W[K][N] f32 row-major -> WT[N][K] f16 row-major (WMMA "B" operand form)
__global__ void k_w2h(const float* __restrict__ W, f16* __restrict__ WT,
                      int K, int Nn) {
  int i = blockIdx.x * blockDim.x + threadIdx.x;
  if (i >= K * Nn) return;
  int k = i / Nn, n = i - k * Nn;
  WT[(size_t)n * K + k] = (f16)W[i];
}

__global__ void k_gather(const float* __restrict__ emb, const int* __restrict__ feat,
                         float* __restrict__ out, size_t E, int D) {
  size_t i = (size_t)blockIdx.x * blockDim.x + threadIdx.x;
  if (i >= E * (size_t)D) return;
  size_t e = i / D;
  int d = (int)(i - e * D);
  out[i] = emb[(size_t)feat[e] * D + d];
}

// mixed = lg + xs[g_src] + xd[g_dst]  (f16 out, feeds GEMMs)
__global__ void k_mixed(const float* __restrict__ lg, const float* __restrict__ xs,
                        const float* __restrict__ xd, const int* __restrict__ gsrc,
                        const int* __restrict__ gdst, f16* __restrict__ out,
                        size_t E, int D) {
  size_t i = (size_t)blockIdx.x * blockDim.x + threadIdx.x;
  if (i >= E * (size_t)D) return;
  size_t e = i / D;
  int d = (int)(i - e * D);
  float v = lg[i] + xs[(size_t)gsrc[e] * D + d] + xd[(size_t)gdst[e] * D + d];
  out[i] = (f16)v;
}

// ------------------------------------------------------------------ GEMM
// C[M][N] = A[M][K] (f16) x BT[N][K] (f16)  + bias + optional residual/relu
// Each wave computes NT 16x16 tiles sharing one A fragment: per k-step
// (1+NT) fragment loads feed NT WMMAs.
template <int NT>
__global__ __launch_bounds__(256) void k_gemm(
    const f16* __restrict__ A, const f16* __restrict__ BT,
    const float* __restrict__ bias, const float* __restrict__ resid,
    float* __restrict__ C32, f16* __restrict__ C16,
    int M, int N, int K, int relu) {
  int lane = threadIdx.x & 31;
  int wave = threadIdx.x >> 5;
  int tn = blockIdx.x * (16 * NT);
  int tm = ((blockIdx.y << 3) + wave) << 4;
  if (tm >= M) return;  // wave-uniform
  v8f zero = {};
  v8f acc[NT];
#pragma unroll
  for (int t = 0; t < NT; ++t) acc[t] = zero;
  for (int k0 = 0; k0 < K; k0 += 32) {
    if (k0 + 32 < K) {  // speculative prefetch of next k-step (gfx1250 path)
      __builtin_prefetch(A + (size_t)tm * K + k0 + 32, 0, 1);
      __builtin_prefetch(BT + (size_t)tn * K + k0 + 32, 0, 1);
    }
    v16h a = load_frag_f16(A, K, tm, k0, lane);
#pragma unroll
    for (int t = 0; t < NT; ++t) {
      v16h b = load_frag_f16(BT, K, tn + 16 * t, k0, lane);
      acc[t] = WMMA(a, b, acc[t]);
    }
  }
  int mb = tm + ((lane >> 4) << 3);
#pragma unroll
  for (int t = 0; t < NT; ++t) {
    int n = tn + 16 * t + (lane & 15);
    float bv = bias ? bias[n] : 0.0f;
#pragma unroll
    for (int r = 0; r < 8; ++r) {
      size_t idx = (size_t)(mb + r) * N + n;
      float val = acc[t][r] + bv;
      if (resid) val += resid[idx];
      if (relu) val = fmaxf(val, 0.0f);
      if (C32) C32[idx] = val;
      if (C16) C16[idx] = (f16)val;
    }
  }
}

// -------------------------------------------------------------- layernorm
// one wave per row
__global__ __launch_bounds__(256) void k_ln(float* __restrict__ X,
                                            const float* __restrict__ g,
                                            const float* __restrict__ b,
                                            f16* __restrict__ X16, int M, int D) {
  int row = blockIdx.x * 8 + (threadIdx.x >> 5);
  int lane = threadIdx.x & 31;
  if (row >= M) return;
  float* xr = X + (size_t)row * D;
  float s = 0.0f;
  for (int d = lane; d < D; d += 32) s += xr[d];
#pragma unroll
  for (int o = 16; o > 0; o >>= 1) s += __shfl_xor(s, o, 32);
  float mean = s / D;
  float vv = 0.0f;
  for (int d = lane; d < D; d += 32) { float t = xr[d] - mean; vv += t * t; }
#pragma unroll
  for (int o = 16; o > 0; o >>= 1) vv += __shfl_xor(vv, o, 32);
  float rstd = rsqrtf(vv / D + 1e-5f);
  for (int d = lane; d < D; d += 32) {
    float y = (xr[d] - mean) * rstd * g[d] + b[d];
    xr[d] = y;
    if (X16) X16[(size_t)row * D + d] = (f16)y;
  }
}

// -------------------------------------------- dense full-graph attention
// one (block b, head h) per workgroup; 128x128 complete graph in LDS.
// S = Q K^T, relation bias via QR = Q rel^T gather, softmax rows,
// out = alpha V + (alpha-histogram W) @ rel
__global__ __launch_bounds__(256) void k_full_attn(
    const float* __restrict__ q, const float* __restrict__ k,
    const float* __restrict__ v, const f16* __restrict__ rel16,
    const f16* __restrict__ relT16, const int* __restrict__ feat,
    float* __restrict__ out) {
  __shared__ f16 Qs[128 * 32];
  __shared__ f16 Ks[128 * 32];
  __shared__ f16 VsT[32 * 128];   // V transposed: [d][src]
  __shared__ f16 Rs[128 * 32];    // rel [r][d]
  __shared__ f16 RTs[32 * 128];   // rel^T [d][r]
  __shared__ float S[128 * 128];  // scores -> alpha (dst-major [j][i])
  __shared__ float W[128 * 128];  // QR [j][r] -> alpha histogram [j][r]

  int b = blockIdx.x, h = blockIdx.y;
  int t = threadIdx.x, lane = t & 31, wave = t >> 5;
  int node0 = b << 7;

  for (int idx = t; idx < 4096; idx += 256) {
    int r = idx >> 5, c = idx & 31;
    size_t off = (((size_t)(node0 + r)) << 8) + (h << 5) + c;  // [N][8][32]
    Qs[idx] = (f16)q[off];
    Ks[idx] = (f16)k[off];
    VsT[c * 128 + r] = (f16)v[off];
    Rs[idx] = rel16[idx];
    RTs[idx] = relT16[idx];
  }
  __syncthreads();

  // S[j][i] = Q[j].K[i] ; W[j][r] = Q[j].rel[r]
  // each wave owns one 16-row j stripe: one A fragment, 16 WMMAs.
  {
    int j0 = wave << 4;
    v16h a = load_frag_f16(Qs, 32, j0, 0, lane);
    int mb = j0 + ((lane >> 4) << 3);
    int nl = lane & 15;
#pragma unroll
    for (int i0 = 0; i0 < 128; i0 += 16) {
      v16h bk = load_frag_f16(Ks, 32, i0, 0, lane);
      v16h br = load_frag_f16(Rs, 32, i0, 0, lane);
      v8f s1 = {}; s1 = WMMA(a, bk, s1);
      v8f s2 = {}; s2 = WMMA(a, br, s2);
      int n = i0 + nl;
#pragma unroll
      for (int r = 0; r < 8; ++r) {
        S[(mb + r) * 128 + n] = s1[r];
        W[(mb + r) * 128 + n] = s2[r];
      }
    }
  }
  __syncthreads();

  // per-dst-row: bias gather + softmax; rebuild W row as alpha histogram
  const float scale = 0.17677669529663687f;  // 1/sqrt(32)
  for (int j = wave; j < 128; j += 8) {
    float sc[4];
    int ff[4];
#pragma unroll
    for (int u = 0; u < 4; ++u) {
      int i = lane + (u << 5);
      int f = feat[(((size_t)b) << 14) + (i << 7) + j];  // edge (b,i->j)
      ff[u] = f;
      sc[u] = (S[j * 128 + i] + W[j * 128 + f]) * scale;
    }
    float mx = fmaxf(fmaxf(sc[0], sc[1]), fmaxf(sc[2], sc[3]));
#pragma unroll
    for (int o = 16; o > 0; o >>= 1) mx = fmaxf(mx, __shfl_xor(mx, o, 32));
    float sum = 0.0f;
#pragma unroll
    for (int u = 0; u < 4; ++u) { sc[u] = __expf(sc[u] - mx); sum += sc[u]; }
#pragma unroll
    for (int o = 16; o > 0; o >>= 1) sum += __shfl_xor(sum, o, 32);
    float inv = 1.0f / sum;
#pragma unroll
    for (int u = 0; u < 4; ++u) W[j * 128 + lane + (u << 5)] = 0.0f;
#pragma unroll
    for (int u = 0; u < 4; ++u) {
      float al = sc[u] * inv;
      S[j * 128 + lane + (u << 5)] = al;
      atomicAdd(&W[j * 128 + ff[u]], al);
    }
  }
  __syncthreads();

  // out[j][d] = sum_i alpha[j][i] V[i][d] + sum_r W[j][r] rel[r][d]
  // wave owns j stripe; A fragments hoisted per k-step, shared by both d tiles.
  {
    int j0 = wave << 4;
    v8f acc0 = {}, acc1 = {};
#pragma unroll
    for (int k0 = 0; k0 < 128; k0 += 32) {
      v16h a1 = load_frag_f32(S, 128, j0, k0, lane);
      v16h a2 = load_frag_f32(W, 128, j0, k0, lane);
      v16h bv0 = load_frag_f16(VsT, 128, 0, k0, lane);
      v16h br0 = load_frag_f16(RTs, 128, 0, k0, lane);
      v16h bv1 = load_frag_f16(VsT, 128, 16, k0, lane);
      v16h br1 = load_frag_f16(RTs, 128, 16, k0, lane);
      acc0 = WMMA(a1, bv0, acc0);
      acc0 = WMMA(a2, br0, acc0);
      acc1 = WMMA(a1, bv1, acc1);
      acc1 = WMMA(a2, br1, acc1);
    }
    int mb = j0 + ((lane >> 4) << 3);
    int nl = lane & 15;
#pragma unroll
    for (int r = 0; r < 8; ++r) {
      size_t orow = (((size_t)(node0 + mb + r)) << 8) + (h << 5);
      out[orow + nl] = acc0[r];
      out[orow + 16 + nl] = acc1[r];
    }
  }
}

// ----------------------------------------------- sparse segment attention
__device__ __forceinline__ void atomicMaxF(float* addr, float val) {
  if (val >= 0.0f)
    atomicMax((int*)addr, __float_as_int(val));
  else
    atomicMin((unsigned int*)addr, __float_as_uint(val));
}

__global__ void k_att_score(const float* __restrict__ q, const float* __restrict__ k,
                            const float* __restrict__ e, const int* __restrict__ src,
                            const int* __restrict__ dst, float* __restrict__ s,
                            float* __restrict__ smax, int E, int Hh, int Dd,
                            int srow, float scale) {
  int idx = blockIdx.x * blockDim.x + threadIdx.x;
  if (idx >= E * Hh) return;
  int eg = idx / Hh, h = idx - eg * Hh;
  int si = src[eg], di = dst[eg];
  const float* kp = k + (size_t)si * srow + h * Dd;
  const float* qp = q + (size_t)di * srow + h * Dd;
  float acc = 0.0f;
  if (e) {
    const float* ep = e + (size_t)eg * Dd;
    for (int d = 0; d < Dd; ++d) acc += (kp[d] + ep[d]) * qp[d];
  } else {
    for (int d = 0; d < Dd; ++d) acc += kp[d] * qp[d];
  }
  acc *= scale;
  s[idx] = acc;
  atomicMaxF(&smax[(size_t)di * Hh + h], acc);
}

__global__ void k_att_norm(const int* __restrict__ dst, float* __restrict__ s,
                           const float* __restrict__ smax, float* __restrict__ den,
                           int E, int Hh) {
  int idx = blockIdx.x * blockDim.x + threadIdx.x;
  if (idx >= E * Hh) return;
  int eg = idx / Hh, h = idx - eg * Hh;
  int di = dst[eg];
  float m = fmaxf(smax[(size_t)di * Hh + h], -1e30f);
  float ex = __expf(s[idx] - m);
  s[idx] = ex;
  atomicAdd(&den[(size_t)di * Hh + h], ex);
}

__global__ void k_att_out(const float* __restrict__ v, const float* __restrict__ e,
                          const int* __restrict__ src, const int* __restrict__ dst,
                          const float* __restrict__ s, const float* __restrict__ den,
                          float* __restrict__ out, int E, int Hh, int Dd,
                          int srow, int orow) {
  int idx = blockIdx.x * blockDim.x + threadIdx.x;
  if (idx >= E * Hh) return;
  int eg = idx / Hh, h = idx - eg * Hh;
  int si = src[eg], di = dst[eg];
  float dn = den[(size_t)di * Hh + h];
  float alpha = s[idx] / (dn > 0.0f ? dn : 1.0f);
  const float* vp = v + (size_t)si * srow + h * Dd;
  float* op = out + (size_t)di * orow + h * Dd;
  if (e) {
    const float* ep = e + (size_t)eg * Dd;
    for (int d = 0; d < Dd; ++d) atomicAdd(&op[d], alpha * (vp[d] + ep[d]));
  } else {
    for (int d = 0; d < Dd; ++d) atomicAdd(&op[d], alpha * vp[d]);
  }
}

// ---------------------------------------------------------------- driver
extern "C" void kernel_launch(void* const* d_in, const int* in_sizes, int n_in,
                              void* d_out, int out_size, void* d_ws, size_t ws_size,
                              hipStream_t stream) {
  (void)in_sizes; (void)n_in; (void)out_size; (void)ws_size;
  const int N = 4096, NDIM = 256, Hn = 8, HHn = 4, DK = 32, EDIM = 32;
  const int Bn = 32, EL = 131072, ELG = 262144, R = 128, Lc = 2;

  const float* in_x     = (const float*)d_in[0];
  const float* rel      = (const float*)d_in[1];
  const int*   g_src    = (const int*)d_in[4];
  const int*   g_dst    = (const int*)d_in[5];
  const int*   lg_src   = (const int*)d_in[6];
  const int*   lg_dst   = (const int*)d_in[7];
  const int*   edge_ft  = (const int*)d_in[8];
  const int*   full_ft  = (const int*)d_in[9];
  const float* nWq = (const float*)d_in[10]; const float* nbq = (const float*)d_in[11];
  const float* nWk = (const float*)d_in[12]; const float* nbk = (const float*)d_in[13];
  const float* nWv = (const float*)d_in[14]; const float* nbv = (const float*)d_in[15];
  const float* nWo = (const float*)d_in[16]; const float* nbo = (const float*)d_in[17];
  const float* nlg = (const float*)d_in[18]; const float* nlb = (const float*)d_in[19];
  const float* nW1 = (const float*)d_in[20]; const float* nb1 = (const float*)d_in[21];
  const float* nW2 = (const float*)d_in[22]; const float* nb2 = (const float*)d_in[23];
  const float* nfg = (const float*)d_in[24]; const float* nfb = (const float*)d_in[25];
  const float* eWs = (const float*)d_in[26]; const float* eWd = (const float*)d_in[27];
  const float* eWq = (const float*)d_in[28]; const float* ebq = (const float*)d_in[29];
  const float* eWk = (const float*)d_in[30]; const float* eWv = (const float*)d_in[31];
  const float* eWo = (const float*)d_in[32]; const float* ebo = (const float*)d_in[33];
  const float* elg = (const float*)d_in[34]; const float* elb = (const float*)d_in[35];
  const float* eW1 = (const float*)d_in[36]; const float* eb1 = (const float*)d_in[37];
  const float* eW2 = (const float*)d_in[38]; const float* eb2 = (const float*)d_in[39];
  const float* efg = (const float*)d_in[40]; const float* efb = (const float*)d_in[41];

  // ---- workspace allocator
  char* wsb = (char*)d_ws;
  size_t off = 0;
  auto alloc = [&](size_t bytes) -> void* {
    void* p = wsb + off;
    off += (bytes + 255) & ~(size_t)255;
    return p;
  };
  float* xa   = (float*)alloc((size_t)N * NDIM * 4);
  float* xb   = (float*)alloc((size_t)N * NDIM * 4);
  float* lga  = (float*)alloc((size_t)EL * EDIM * 4);
  float* lgb  = (float*)alloc((size_t)EL * EDIM * 4);
  float* qb   = (float*)alloc((size_t)N * NDIM * 4);
  float* kbuf = (float*)alloc((size_t)N * NDIM * 4);
  float* vbuf = (float*)alloc((size_t)N * NDIM * 4);
  float* obuf = (float*)alloc((size_t)N * NDIM * 4);
  float* xs   = (float*)alloc((size_t)N * EDIM * 4);
  float* xd   = (float*)alloc((size_t)N * EDIM * 4);
  float* eqb  = (float*)alloc((size_t)EL * EDIM * 4);
  float* ekb  = (float*)alloc((size_t)EL * EDIM * 4);
  float* evb  = (float*)alloc((size_t)EL * EDIM * 4);
  float* eob  = (float*)alloc((size_t)EL * EDIM * 4);
  float* sbuf = (float*)alloc((size_t)ELG * Hn * 4);
  float* smax = (float*)alloc((size_t)EL * Hn * 4);
  float* den  = (float*)alloc((size_t)EL * Hn * 4);
  f16* x16a   = (f16*)alloc((size_t)N * NDIM * 2);
  f16* x16b   = (f16*)alloc((size_t)N * NDIM * 2);
  f16* lg16   = (f16*)alloc((size_t)EL * EDIM * 2);
  f16* o16    = (f16*)alloc((size_t)N * NDIM * 2);
  f16* h116   = (f16*)alloc((size_t)N * 4 * NDIM * 2);
  f16* mix16  = (f16*)alloc((size_t)EL * EDIM * 2);
  f16* eo16   = (f16*)alloc((size_t)EL * EDIM * 2);
  f16* eh116  = (f16*)alloc((size_t)EL * 4 * EDIM * 2);
  f16* rel16  = (f16*)alloc((size_t)R * EDIM * 2);
  f16* relT16 = (f16*)alloc((size_t)R * EDIM * 2);
  f16 *WqT[2], *WkT[2], *WvT[2], *WoT[2], *W1T[2], *W2T[2];
  f16 *WsT[2], *WdT[2], *eWqT[2], *eWkT[2], *eWvT[2], *eWoT[2], *eW1T[2], *eW2T[2];
  for (int l = 0; l < Lc; ++l) {
    WqT[l]  = (f16*)alloc(65536 * 2);  WkT[l]  = (f16*)alloc(65536 * 2);
    WvT[l]  = (f16*)alloc(65536 * 2);  WoT[l]  = (f16*)alloc(65536 * 2);
    W1T[l]  = (f16*)alloc(262144 * 2); W2T[l]  = (f16*)alloc(262144 * 2);
    WsT[l]  = (f16*)alloc(8192 * 2);   WdT[l]  = (f16*)alloc(8192 * 2);
    eWqT[l] = (f16*)alloc(1024 * 2);   eWkT[l] = (f16*)alloc(1024 * 2);
    eWvT[l] = (f16*)alloc(1024 * 2);   eWoT[l] = (f16*)alloc(1024 * 2);
    eW1T[l] = (f16*)alloc(4096 * 2);   eW2T[l] = (f16*)alloc(4096 * 2);
  }

  auto gx = [](size_t n) { return dim3((unsigned)((n + 255) / 256)); };
  auto w2h = [&](const float* W, f16* WT, int K, int Nn) {
    k_w2h<<<gx((size_t)K * Nn), dim3(256), 0, stream>>>(W, WT, K, Nn);
  };
  auto fillk = [&](float* p, float v, size_t n) {
    k_fill<<<gx(n), dim3(256), 0, stream>>>(p, v, n);
  };
  auto cvt = [&](const float* in, f16* out, size_t n) {
    k_cvt<<<gx(n), dim3(256), 0, stream>>>(in, out, n);
  };
  auto gemm = [&](const f16* A, const f16* BT, const float* bias, const float* res,
                  float* C32, f16* C16, int M, int Nn, int K, int relu) {
    if (Nn % 64 == 0) {
      dim3 grid((unsigned)(Nn / 64), (unsigned)(M / 128));
      k_gemm<4><<<grid, dim3(256), 0, stream>>>(A, BT, bias, res, C32, C16, M, Nn, K, relu);
    } else {
      dim3 grid((unsigned)(Nn / 32), (unsigned)(M / 128));
      k_gemm<2><<<grid, dim3(256), 0, stream>>>(A, BT, bias, res, C32, C16, M, Nn, K, relu);
    }
  };

  // ---- one-time prep
  for (int l = 0; l < Lc; ++l) {
    size_t wo = (size_t)l * NDIM * NDIM;
    w2h(nWq + wo, WqT[l], NDIM, NDIM);
    w2h(nWk + wo, WkT[l], NDIM, NDIM);
    w2h(nWv + wo, WvT[l], NDIM, NDIM);
    w2h(nWo + wo, WoT[l], NDIM, NDIM);
    w2h(nW1 + (size_t)l * NDIM * 4 * NDIM, W1T[l], NDIM, 4 * NDIM);
    w2h(nW2 + (size_t)l * 4 * NDIM * NDIM, W2T[l], 4 * NDIM, NDIM);
    w2h(eWs + (size_t)l * NDIM * EDIM, WsT[l], NDIM, EDIM);
    w2h(eWd + (size_t)l * NDIM * EDIM, WdT[l], NDIM, EDIM);
    w2h(eWq + (size_t)l * EDIM * EDIM, eWqT[l], EDIM, EDIM);
    w2h(eWk + (size_t)l * EDIM * EDIM, eWkT[l], EDIM, EDIM);
    w2h(eWv + (size_t)l * EDIM * EDIM, eWvT[l], EDIM, EDIM);
    w2h(eWo + (size_t)l * EDIM * EDIM, eWoT[l], EDIM, EDIM);
    w2h(eW1 + (size_t)l * EDIM * 4 * EDIM, eW1T[l], EDIM, 4 * EDIM);
    w2h(eW2 + (size_t)l * 4 * EDIM * EDIM, eW2T[l], 4 * EDIM, EDIM);
  }
  cvt(rel, rel16, (size_t)R * EDIM);
  w2h(rel, relT16, R, EDIM);
  hipMemcpyAsync(xa, in_x, (size_t)N * NDIM * 4, hipMemcpyDeviceToDevice, stream);
  cvt(in_x, x16a, (size_t)N * NDIM);
  k_gather<<<gx((size_t)EL * EDIM), dim3(256), 0, stream>>>(rel, edge_ft, lga,
                                                            (size_t)EL, EDIM);

  const float neg_inf = -__builtin_inff();
  const float sc32 = 0.17677669529663687f;  // 1/sqrt(32)

  for (int l = 0; l < Lc; ++l) {
    // ---------- node update ----------
    gemm(x16a, WqT[l], nbq + (size_t)l * NDIM, nullptr, qb, nullptr, N, NDIM, NDIM, 0);
    gemm(x16a, WkT[l], nbk + (size_t)l * NDIM, nullptr, kbuf, nullptr, N, NDIM, NDIM, 0);
    gemm(x16a, WvT[l], nbv + (size_t)l * NDIM, nullptr, vbuf, nullptr, N, NDIM, NDIM, 0);
    fillk(obuf, 0.0f, (size_t)N * NDIM);
    k_full_attn<<<dim3(Bn, HHn), dim3(256), 0, stream>>>(qb, kbuf, vbuf, rel16,
                                                         relT16, full_ft, obuf);
    // local sparse heads 4..7 (relation bias = lg)
    fillk(smax, neg_inf, (size_t)N * HHn);
    fillk(den, 0.0f, (size_t)N * HHn);
    int E1 = EL * HHn;
    k_att_score<<<gx(E1), dim3(256), 0, stream>>>(qb + HHn * DK, kbuf + HHn * DK, lga,
                                                  g_src, g_dst, sbuf, smax, EL, HHn,
                                                  DK, NDIM, sc32);
    k_att_norm<<<gx(E1), dim3(256), 0, stream>>>(g_dst, sbuf, smax, den, EL, HHn);
    k_att_out<<<gx(E1), dim3(256), 0, stream>>>(vbuf + HHn * DK, lga, g_src, g_dst,
                                                sbuf, den, obuf + HHn * DK, EL, HHn,
                                                DK, NDIM, NDIM);
    cvt(obuf, o16, (size_t)N * NDIM);
    gemm(o16, WoT[l], nbo + (size_t)l * NDIM, xa, xb, nullptr, N, NDIM, NDIM, 0);
    k_ln<<<dim3(N / 8), dim3(256), 0, stream>>>(xb, nlg + (size_t)l * NDIM,
                                                nlb + (size_t)l * NDIM, x16b, N, NDIM);
    gemm(x16b, W1T[l], nb1 + (size_t)l * 4 * NDIM, nullptr, nullptr, h116,
         N, 4 * NDIM, NDIM, 1);
    gemm(h116, W2T[l], nb2 + (size_t)l * NDIM, xb, xb, nullptr, N, NDIM, 4 * NDIM, 0);
    k_ln<<<dim3(N / 8), dim3(256), 0, stream>>>(xb, nfg + (size_t)l * NDIM,
                                                nfb + (size_t)l * NDIM, x16b, N, NDIM);

    // ---------- edge update (uses PRE-update x = x16a) ----------
    gemm(x16a, WsT[l], nullptr, nullptr, xs, nullptr, N, EDIM, NDIM, 0);
    gemm(x16a, WdT[l], nullptr, nullptr, xd, nullptr, N, EDIM, NDIM, 0);
    k_mixed<<<gx((size_t)EL * EDIM), dim3(256), 0, stream>>>(lga, xs, xd, g_src,
                                                             g_dst, mix16,
                                                             (size_t)EL, EDIM);
    gemm(mix16, eWqT[l], ebq + (size_t)l * EDIM, nullptr, eqb, nullptr, EL, EDIM, EDIM, 0);
    gemm(mix16, eWkT[l], nullptr, nullptr, ekb, nullptr, EL, EDIM, EDIM, 0);
    gemm(mix16, eWvT[l], nullptr, nullptr, evb, nullptr, EL, EDIM, EDIM, 0);
    fillk(smax, neg_inf, (size_t)EL * Hn);
    fillk(den, 0.0f, (size_t)EL * Hn);
    fillk(eob, 0.0f, (size_t)EL * EDIM);
    int E2 = ELG * Hn;
    k_att_score<<<gx(E2), dim3(256), 0, stream>>>(eqb, ekb, nullptr, lg_src, lg_dst,
                                                  sbuf, smax, ELG, Hn, 4, EDIM, 0.5f);
    k_att_norm<<<gx(E2), dim3(256), 0, stream>>>(lg_dst, sbuf, smax, den, ELG, Hn);
    k_att_out<<<gx(E2), dim3(256), 0, stream>>>(evb, nullptr, lg_src, lg_dst, sbuf,
                                                den, eob, ELG, Hn, 4, EDIM, EDIM);
    cvt(eob, eo16, (size_t)EL * EDIM);
    gemm(eo16, eWoT[l], ebo + (size_t)l * EDIM, lga, lgb, nullptr, EL, EDIM, EDIM, 0);
    k_ln<<<dim3(EL / 8), dim3(256), 0, stream>>>(lgb, elg + (size_t)l * EDIM,
                                                 elb + (size_t)l * EDIM, lg16, EL, EDIM);
    gemm(lg16, eW1T[l], eb1 + (size_t)l * 4 * EDIM, nullptr, nullptr, eh116,
         EL, 4 * EDIM, EDIM, 1);
    gemm(eh116, eW2T[l], eb2 + (size_t)l * EDIM, lgb, lgb, nullptr, EL, EDIM, 4 * EDIM, 0);
    k_ln<<<dim3(EL / 8), dim3(256), 0, stream>>>(lgb, efg + (size_t)l * EDIM,
                                                 efb + (size_t)l * EDIM, nullptr, EL, EDIM);

    { float* t = xa; xa = xb; xb = t; }
    { f16* t = x16a; x16a = x16b; x16b = t; }
    { float* t = lga; lga = lgb; lgb = t; }
  }

  hipMemcpyAsync(d_out, xa, (size_t)N * NDIM * 4, hipMemcpyDeviceToDevice, stream);
  hipMemcpyAsync((float*)d_out + (size_t)N * NDIM, lga, (size_t)EL * EDIM * 4,
                 hipMemcpyDeviceToDevice, stream);
}
